// SimpleGRU_89807766159401
// MI455X (gfx1250) — matrix operand order repaired and checked by reading, not compiled, not run on hardware
//
#include <hip/hip_runtime.h>

// ---------------------------------------------------------------------------
// 2-layer GRU (T=15, B=131072, IN=18, H=32) + mean-pool + 2-class linear.
// One wave32 = one 16-batch tile across all timesteps.
// Matmuls: v_wmma_f32_16x16x32_f16 (K=32 == H), 24 WMMAs / step / wave.
// All weight B-fragments pre-packed f16 in LDS; Whh0/Whh1 loaded once into
// VGPRs (recurrent critical path), Wih0/Wih1 streamed per use (2x ds_load_b128).
// Activations use v_exp_f32/v_rcp_f32 (TRANS pipe, co-executes with VALU).
// Layers software-pipelined: L1(t) and L0(t+1) are independent WMMA chains.
// ---------------------------------------------------------------------------

#define T_STEPS 15
#define BATCH   131072
#define IN_DIM  18
#define HDIM    32
#define NCLS    2
#define WAVES_PER_BLOCK 8     // 256 threads, 128 batch rows per block
#define HROW    48            // padded row stride (halfs) for h staging

typedef __attribute__((ext_vector_type(16))) _Float16 v16h;
typedef __attribute__((ext_vector_type(8)))  _Float16 v8h;
typedef __attribute__((ext_vector_type(8)))  float    v8f;

struct f2 { float x, y; };

#define LOG2E 1.442695040888963f

static __device__ __forceinline__ v8f wmma_f16(v16h a, v16h b, v8f c) {
    return __builtin_amdgcn_wmma_f32_16x16x32_f16(
        false, a, false, b, (short)0, c, false, false);
}

static __device__ __forceinline__ v8f splat8(float v) {
    v8f c;
#pragma unroll
    for (int i = 0; i < 8; ++i) c[i] = v;
    return c;
}

// sigmoid(x + b) with bs = -LOG2E*b precomputed: 1/(1 + 2^(-LOG2E*x + bs))
static __device__ __forceinline__ float sigm_b(float x, float bs) {
    float e = __builtin_amdgcn_exp2f(fmaf(-LOG2E, x, bs));
    return __builtin_amdgcn_rcpf(1.0f + e);
}
// tanh(p) = 1 - 2/(1 + 2^(2*LOG2E*p)); saturates cleanly at +-1
static __device__ __forceinline__ float tanh_fast(float p) {
    float e = __builtin_amdgcn_exp2f((2.0f * LOG2E) * p);
    return fmaf(-2.0f, __builtin_amdgcn_rcpf(1.0f + e), 1.0f);
}

// ---- B-matrix fragment (32x16 f16) from global weight W[3H][K] ------------
// Lane l, dword v holds pair W[n][k],W[n][k+1]; n = 16j + l16, k = 2(v&3) + 8hi + 16(v>=4).
template <int K>
static __device__ __forceinline__ v16h load_wfrag(const float* __restrict__ W,
                                                  int j, int l16, int hi) {
    v16h b;
    const float* row = W + (size_t)(j * 16 + l16) * K;
#pragma unroll
    for (int v = 0; v < 8; ++v) {
        int k = 2 * (v & 3) + 8 * hi + ((v >= 4) ? 16 : 0);
        float fx = 0.0f, fy = 0.0f;
        if (k + 1 < K) { f2 p = *(const f2*)(row + k); fx = p.x; fy = p.y; }
        b[2 * v]     = (_Float16)fx;
        b[2 * v + 1] = (_Float16)fy;
    }
    return b;
}

// ---- A-matrix fragment (16x32 f16) from global x (K=18, zero-padded) ------
static __device__ __forceinline__ v16h load_xfrag(const float* __restrict__ xrow0,
                                                  int l16, int hi) {
    const float* row = xrow0 + (size_t)l16 * IN_DIM;
    v16h a;
#pragma unroll
    for (int v = 0; v < 8; ++v) {
        int k = 2 * v + 8 * hi + ((v >= 4) ? 8 : 0);
        float fx = 0.0f, fy = 0.0f;
        if (k + 1 < IN_DIM) { f2 p = *(const f2*)(row + k); fx = p.x; fy = p.y; }
        a[2 * v]     = (_Float16)fx;
        a[2 * v + 1] = (_Float16)fy;
    }
    return a;
}

// ---- LDS-resident pre-packed fragment: two ds_load_b128 (16B lane stride) -
static __device__ __forceinline__ v16h ld_wfrag_lds(const _Float16* __restrict__ base,
                                                    int f, int lane) {
    const _Float16* p = base + (size_t)f * (2 * 32 * 8);
    const v8h lo = *(const v8h*)(p + (0 * 32 + lane) * 8);
    const v8h hh = *(const v8h*)(p + (1 * 32 + lane) * 8);
    v16h a;
#pragma unroll
    for (int i = 0; i < 8; ++i) { a[i] = lo[i]; a[8 + i] = hh[i]; }
    return a;
}

// ---- C-layout (2 x v8f) -> LDS row-major f16, padded rows -----------------
static __device__ __forceinline__ void stage_h(_Float16* __restrict__ hs,
                                               const v8f* h, int l16, int hi) {
#pragma unroll
    for (int tile = 0; tile < 2; ++tile)
#pragma unroll
        for (int r = 0; r < 8; ++r)
            hs[(r + 8 * hi) * HROW + l16 + 16 * tile] = (_Float16)h[tile][r];
}

// ---- LDS row-major f16 -> A-matrix fragment (two ds_load_b128) ------------
static __device__ __forceinline__ v16h load_hfrag(const _Float16* __restrict__ hs,
                                                  int l16, int hi) {
    const v8h lo = *(const v8h*)(hs + l16 * HROW + 8 * hi);
    const v8h hh = *(const v8h*)(hs + l16 * HROW + 16 + 8 * hi);
    v16h a;
#pragma unroll
    for (int i = 0; i < 8; ++i) { a[i] = lo[i]; a[8 + i] = hh[i]; }
    return a;
}

// ---- GRU cell for one 16-batch tile (gate order r,z,n) --------------------
// wih streamed from LDS; whh in VGPRs; biases folded into activation FMAs:
// lb[0..63]=bih+bhh (r,z), lb[64..95]=bih_n, lb[96..127]=bhh_n.
static __device__ __forceinline__ void gru_cell(const v16h& ax, const v16h& ah,
                                                const _Float16* __restrict__ wfl,
                                                const v16h (&whh)[6],
                                                const float* __restrict__ lb,
                                                const v8f& czero,
                                                int lane, int l16, v8f (&h)[2]) {
#pragma unroll
    for (int hb = 0; hb < 2; ++hb) {
        v8f cr = wmma_f16(ax, ld_wfrag_lds(wfl, hb, lane), czero);
        cr = wmma_f16(ah, whh[hb], cr);
        v8f cz = wmma_f16(ax, ld_wfrag_lds(wfl, 2 + hb, lane), czero);
        cz = wmma_f16(ah, whh[2 + hb], cz);
        v8f cn = wmma_f16(ax, ld_wfrag_lds(wfl, 4 + hb, lane), czero);
        v8f ch = wmma_f16(ah, whh[4 + hb], czero);

        const float brs = -LOG2E * lb[16 * hb + l16];
        const float bzs = -LOG2E * lb[16 * (2 + hb) + l16];
        const float bin = lb[64 + 16 * hb + l16];
        const float bhn = lb[96 + 16 * hb + l16];
#pragma unroll
        for (int i = 0; i < 8; ++i) {
            float r = sigm_b(cr[i], brs);
            float z = sigm_b(cz[i], bzs);
            float n = tanh_fast(fmaf(r, ch[i] + bhn, cn[i] + bin));
            h[hb][i] = fmaf(z, h[hb][i] - n, n);   // (1-z)*n + z*h
        }
    }
}

__global__ __launch_bounds__(256) void
gru2_wmma_kernel(const float* __restrict__ x,
                 const float* __restrict__ Wih0, const float* __restrict__ Whh0,
                 const float* __restrict__ bih0, const float* __restrict__ bhh0,
                 const float* __restrict__ Wih1, const float* __restrict__ Whh1,
                 const float* __restrict__ bih1, const float* __restrict__ bhh1,
                 const float* __restrict__ Wout, const float* __restrict__ bout,
                 float* __restrict__ out) {
    // all 24 pre-packed weight fragments: [frag][half][lane][8 halfs] = 24 KB
    // 0..5 Wih0, 6..11 Wih1, 12..17 Whh0, 18..23 Whh1
    __shared__ __align__(16) _Float16 wfrags[24][2][32][8];
    // per-wave h staging (two buffers, padded rows); reused for f32 pool dump
    __shared__ __align__(16) _Float16 stg[WAVES_PER_BLOCK][2][16 * HROW];   // 24 KB
    __shared__ float lbias[2][128];                                         // 1 KB

    const int lane = threadIdx.x & 31;
    const int wave = threadIdx.x >> 5;
    const int l16  = lane & 15;
    const int hi   = lane >> 4;
    const int b0   = (blockIdx.x * WAVES_PER_BLOCK + wave) * 16;

    // ---- cooperative setup: biases + all weight fragments into LDS --------
    if (threadIdx.x < 64) {
        int i = threadIdx.x;
        lbias[0][i] = bih0[i] + bhh0[i];
        lbias[1][i] = bih1[i] + bhh1[i];
        if (i < 32) {
            lbias[0][64 + i] = bih0[64 + i];  lbias[0][96 + i] = bhh0[64 + i];
            lbias[1][64 + i] = bih1[64 + i];  lbias[1][96 + i] = bhh1[64 + i];
        }
    }
    for (int f = wave; f < 24; f += WAVES_PER_BLOCK) {
        const int M = f / 6, j = f % 6;
        v16h b;
        if      (M == 0) b = load_wfrag<IN_DIM>(Wih0, j, l16, hi);
        else if (M == 1) b = load_wfrag<HDIM>(Wih1, j, l16, hi);
        else if (M == 2) b = load_wfrag<HDIM>(Whh0, j, l16, hi);
        else             b = load_wfrag<HDIM>(Whh1, j, l16, hi);
        v8h lo, hh;
#pragma unroll
        for (int i = 0; i < 8; ++i) { lo[i] = b[i]; hh[i] = b[8 + i]; }
        *(v8h*)&wfrags[f][0][lane][0] = lo;
        *(v8h*)&wfrags[f][1][lane][0] = hh;
    }
    __syncthreads();
    const _Float16* wfb = &wfrags[0][0][0][0];
    const _Float16* wf0 = wfb;                       // Wih0 frags
    const _Float16* wf1 = &wfrags[6][0][0][0];       // Wih1 frags

    // ---- recurrent weights resident in VGPRs (f16 loads, not remat-able) --
    v16h whh0f[6], whh1f[6];
#pragma unroll
    for (int j = 0; j < 6; ++j) {
        whh0f[j] = ld_wfrag_lds(wfb, 12 + j, lane);
        whh1f[j] = ld_wfrag_lds(wfb, 18 + j, lane);
    }

    v8f h0[2], h1[2], pool[2];
    const v8f czero = splat8(0.0f);
#pragma unroll
    for (int tl = 0; tl < 2; ++tl) { h0[tl] = czero; h1[tl] = czero; pool[tl] = czero; }
    v16h zfrag;
#pragma unroll
    for (int i = 0; i < 16; ++i) zfrag[i] = (_Float16)0.0f;

    _Float16* hsA = &stg[wave][0][0];
    _Float16* hsB = &stg[wave][1][0];

    // ---- prologue: layer-0 step 0 (h = 0) ---------------------------------
    v16h ax = load_xfrag(x + (size_t)b0 * IN_DIM, l16, hi);
    gru_cell(ax, zfrag, wf0, whh0f, lbias[0], czero, lane, l16, h0);
    v16h aH1 = zfrag;

    // ---- pipelined time loop: L1(t) and L0(t+1) are independent chains ----
    for (int t = 0; t < T_STEPS - 1; ++t) {
        stage_h(hsA, h0, l16, hi);
        asm volatile("s_wait_dscnt 0" ::: "memory");
        v16h a1 = load_hfrag(hsA, l16, hi);        // y1_t == h0_t
        asm volatile("" ::: "memory");

        ax = load_xfrag(x + ((size_t)(t + 1) * BATCH + b0) * IN_DIM, l16, hi);
        gru_cell(ax, a1, wf0, whh0f, lbias[0], czero, lane, l16, h0);  // h0_{t+1}
        gru_cell(a1, aH1, wf1, whh1f, lbias[1], czero, lane, l16, h1); // h1_t

        stage_h(hsB, h1, l16, hi);
        asm volatile("s_wait_dscnt 0" ::: "memory");
        aH1 = load_hfrag(hsB, l16, hi);
        asm volatile("" ::: "memory");
#pragma unroll
        for (int tl = 0; tl < 2; ++tl) pool[tl] += h1[tl];
    }
    // ---- epilogue: layer-1 step T-1 ---------------------------------------
    {
        stage_h(hsA, h0, l16, hi);
        asm volatile("s_wait_dscnt 0" ::: "memory");
        v16h a1 = load_hfrag(hsA, l16, hi);
        asm volatile("" ::: "memory");
        gru_cell(a1, aH1, wf1, whh1f, lbias[1], czero, lane, l16, h1);
#pragma unroll
        for (int tl = 0; tl < 2; ++tl) pool[tl] += h1[tl];
    }

    // ---- mean-pool + 2-class linear: 32 lanes = 16 rows x 2 classes -------
    float* ps = (float*)&stg[wave][0][0];          // 2 KB fits in 3 KB region
    asm volatile("" ::: "memory");
#pragma unroll
    for (int tile = 0; tile < 2; ++tile)
#pragma unroll
        for (int r = 0; r < 8; ++r)
            ps[(r + 8 * hi) * HDIM + l16 + 16 * tile] = pool[tile][r];
    asm volatile("s_wait_dscnt 0" ::: "memory");

    const int m = l16, c = hi;
    float s = 0.0f;
#pragma unroll
    for (int n = 0; n < HDIM; ++n)
        s = fmaf(ps[m * HDIM + n], Wout[c * HDIM + n], s);
    out[(size_t)(b0 + m) * NCLS + c] = fmaf(s, 1.0f / (float)T_STEPS, bout[c]);
}

extern "C" void kernel_launch(void* const* d_in, const int* in_sizes, int n_in,
                              void* d_out, int out_size, void* d_ws, size_t ws_size,
                              hipStream_t stream) {
    (void)in_sizes; (void)n_in; (void)out_size; (void)d_ws; (void)ws_size;
    const float* x    = (const float*)d_in[0];
    const float* Wih0 = (const float*)d_in[1];
    const float* Whh0 = (const float*)d_in[2];
    const float* bih0 = (const float*)d_in[3];
    const float* bhh0 = (const float*)d_in[4];
    const float* Wih1 = (const float*)d_in[5];
    const float* Whh1 = (const float*)d_in[6];
    const float* bih1 = (const float*)d_in[7];
    const float* bhh1 = (const float*)d_in[8];
    const float* Wout = (const float*)d_in[9];
    const float* bout = (const float*)d_in[10];
    float* out = (float*)d_out;

    dim3 grid(BATCH / (16 * WAVES_PER_BLOCK));   // 1024 blocks
    dim3 block(32 * WAVES_PER_BLOCK);            // 256 threads = 8 waves
    gru2_wmma_kernel<<<grid, block, 0, stream>>>(x, Wih0, Whh0, bih0, bhh0,
                                                 Wih1, Whh1, bih1, bhh1,
                                                 Wout, bout, out);
}
